// ReprogrammingLayer_43757126811740
// MI455X (gfx1250) — matrix operand order repaired and checked by reading, not compiled, not run on hardware
//
#include <hip/hip_runtime.h>

// ---------------------------------------------------------------------------
// CDNA5 / gfx1250 cross-attention (ReprogrammingLayer), bf16 WMMA + fp32 acc.
// B=32, L=1024, S=1000 (padded 1024), D_MODEL=512, D_LLM=4096, H=8, E=16.
// Round 2: hardware bf16 conversion + prepacked (transposed bf16) weights.
// ---------------------------------------------------------------------------

typedef __attribute__((ext_vector_type(16))) __bf16 v16bf;
typedef __attribute__((ext_vector_type(8)))  float  v8f;

union BF16x16 { v16bf v; __bf16 h[16]; unsigned short u[16]; unsigned int d[8]; };
union F32x8   { v8f   v; float f[8]; };

__device__ __forceinline__ __bf16 f2bf(float x) { return (__bf16)x; }  // hw cvt

__device__ __forceinline__ v8f wmma_bf16(const BF16x16& a, const BF16x16& b, v8f c) {
    return __builtin_amdgcn_wmma_f32_16x16x32_bf16(
        /*neg_a=*/false, a.v, /*neg_b=*/false, b.v,
        /*c_mod=*/(short)0, c, /*reuse_a=*/false, /*reuse_b=*/false);
}

// A-matrix (16x32 bf16) element j of lane (half,m):  K = (j<8) ? half*8+j : 16+half*8+(j-8)
// B-matrix (32x16 bf16) element e of lane (half,n):  K in A's order (gemm_tile: half*16+e for A and B), N = n
// C/D (16x16 f32)       vgpr  r of lane (half,n):    M = r + half*8, N = n

// ------------------------- weight prepack (transpose) ----------------------
// Wt[n][k] = bf16(W[k][n]); makes the WMMA B-operand a contiguous 32B lane load.
__global__ __launch_bounds__(256) void prepack_w_k(const float* __restrict__ W,
                                                   unsigned short* __restrict__ Wt,
                                                   int K, int N) {
    int idx = blockIdx.x * 256 + threadIdx.x;     // over N*K
    if (idx >= K * N) return;
    int n = idx / K, k = idx - n * K;
    BF16x16 tmp; tmp.h[0] = f2bf(W[(size_t)k * N + n]);
    Wt[idx] = tmp.u[0];
}

// 16x16-per-wave GEMM tile: A fp32 (hw-converted), B prepacked bf16 [N][K].
template <int KTOT>
__device__ __forceinline__ v8f gemm_tile(const float* __restrict__ A, int lda,
                                         bool rowValid, size_t arow,
                                         const unsigned short* __restrict__ Bt, int bcol) {
    int lane = threadIdx.x & 31, half = lane >> 4;
    v8f acc = {};
    const unsigned short* bp = Bt + (size_t)bcol * KTOT + half * 16;
    for (int k0 = 0; k0 < KTOT; k0 += 32) {
        BF16x16 a, b;
        if (rowValid) {
            const float* ap = A + arow * (size_t)lda + k0 + half * 16;
            #pragma unroll
            for (int j = 0; j < 8; ++j) a.h[j]     = f2bf(ap[j]);        // K = half*16 + j
            #pragma unroll
            for (int j = 0; j < 8; ++j) a.h[8 + j] = f2bf(ap[8 + j]);    // K = half*16+8+j
        } else {
            #pragma unroll
            for (int j = 0; j < 8; ++j) a.d[j] = 0;
        }
        *(uint4*)&b.d[0] = ((const uint4*)(bp + k0))[0];                 // K=half*16+e
        *(uint4*)&b.d[4] = ((const uint4*)(bp + k0))[1];
        acc = wmma_bf16(a, b, acc);
    }
    return acc;
}

// --------------------------- K / V projections -----------------------------
// K stored as Kb[h][s][e] (bf16, s padded to 1024); V stored transposed
// Vt[h][e][s] so the attention P·V B-operand is two 16B lane loads.
__global__ __launch_bounds__(128) void proj_kv_k(
    const float* __restrict__ src, const float* __restrict__ val,
    const unsigned short* __restrict__ WkT, const float* __restrict__ bk,
    const unsigned short* __restrict__ WvT, const float* __restrict__ bv,
    unsigned short* __restrict__ Kb, unsigned short* __restrict__ Vt) {
    int lane = threadIdx.x & 31, wave = threadIdx.x >> 5;
    int half = lane >> 4, lm = lane & 15;
    int tile  = blockIdx.x * 4 + wave;      // 0..1023
    int which = tile >> 9;                  // 0 = K, 1 = V
    int t     = tile & 511;
    int mt    = t >> 3, h = t & 7;

    const float*          A    = which ? val : src;
    const unsigned short* Bt   = which ? WvT : WkT;
    const float*          bias = which ? bv  : bk;

    int row = mt * 16 + lm;
    v8f acc = gemm_tile<4096>(A, 4096, row < 1000, (size_t)row, Bt, h * 16 + lm);

    F32x8 d; d.v = acc;
    float bb = bias[h * 16 + lm];
    BF16x16 cv;
    #pragma unroll
    for (int r = 0; r < 8; ++r) {
        int srow = mt * 16 + r + half * 8;
        cv.h[0] = f2bf(d.f[r] + bb);        // padded rows masked later in softmax
        if (which == 0)
            Kb[((size_t)h * 1024 + srow) * 16 + lm] = cv.u[0];
        else
            Vt[((size_t)h * 16 + lm) * 1024 + srow] = cv.u[0];
    }
}

// ------------------------------ Q projection -------------------------------
__global__ __launch_bounds__(128) void proj_q_k(
    const float* __restrict__ tgt, const unsigned short* __restrict__ WqT,
    const float* __restrict__ bq, unsigned short* __restrict__ Qb) {
    int lane = threadIdx.x & 31, wave = threadIdx.x >> 5;
    int half = lane >> 4, lm = lane & 15;
    int tile = blockIdx.x * 4 + wave;       // 0..16383
    int mt   = tile >> 3, h = tile & 7;
    int row  = mt * 16 + lm;

    v8f acc = gemm_tile<512>(tgt, 512, true, (size_t)row, WqT, h * 16 + lm);
    F32x8 d; d.v = acc;
    float bb = bq[h * 16 + lm];
    BF16x16 cv;
    #pragma unroll
    for (int r = 0; r < 8; ++r) {
        cv.h[0] = f2bf(d.f[r] + bb);
        Qb[((size_t)mt * 16 + r + half * 8) * 128 + h * 16 + lm] = cv.u[0];
    }
}

// --------------------------- flash attention core --------------------------
__global__ __launch_bounds__(128) void attn_k(
    const unsigned short* __restrict__ Qb, const unsigned short* __restrict__ Kb,
    const unsigned short* __restrict__ Vt, unsigned short* __restrict__ Ob) {
    __shared__ unsigned short Pex[4][16][32];   // per-wave P tile, D-layout -> A-layout
    int lane = threadIdx.x & 31, wave = threadIdx.x >> 5;
    int half = lane >> 4, lm = lane & 15;

    int bi = blockIdx.x;                    // 32*8*16 = 4096 blocks
    int lt = (bi & 15) * 4 + wave;          // L-tile 0..63
    int h  = (bi >> 4) & 7;
    int b  = bi >> 7;
    int l0 = lt * 16;

    // Q as A-matrix, K-dim padded 16->32 with zeros (upper half).
    BF16x16 aq;
    {
        const unsigned short* qp =
            Qb + ((size_t)(b * 1024 + l0 + lm)) * 128 + h * 16;
        if (half == 0) {
            *(uint4*)&aq.d[0] = ((const uint4*)qp)[0];
            *(uint4*)&aq.d[4] = ((const uint4*)qp)[1];
        } else {
            #pragma unroll
            for (int j = 0; j < 8; ++j) aq.d[j] = 0;
        }
    }

    float m[8], lsum[8];
    F32x8 o;
    #pragma unroll
    for (int r = 0; r < 8; ++r) { m[r] = -3.0e38f; lsum[r] = 0.f; o.f[r] = 0.f; }

    const float scale = 0.25f;              // 1/sqrt(16)
    for (int s0 = 0; s0 < 1024; s0 += 32) {
        // K^T as B-matrix: rows 0..15 = E (valid, half 0), rows 16..31 zero.
        BF16x16 bk0, bk1;
        if (half == 0) {
            const unsigned short* kp = Kb + ((size_t)h * 1024 + s0 + lm) * 16;
            *(uint4*)&bk0.d[0] = ((const uint4*)kp)[0];
            *(uint4*)&bk0.d[4] = ((const uint4*)kp)[1];
            const unsigned short* kp1 = kp + 256;   // +16 S rows
            *(uint4*)&bk1.d[0] = ((const uint4*)kp1)[0];
            *(uint4*)&bk1.d[4] = ((const uint4*)kp1)[1];
        } else {
            #pragma unroll
            for (int j = 0; j < 8; ++j) { bk0.d[j] = 0; bk1.d[j] = 0; }
        }
        v8f zero = {};
        F32x8 sc0, sc1;
        sc0.v = wmma_bf16(aq, bk0, zero);
        sc1.v = wmma_bf16(aq, bk1, zero);

        // Online softmax; row r+half*8 stats shared across the 16-lane group.
        float p0[8], p1[8];
        #pragma unroll
        for (int r = 0; r < 8; ++r) {
            float x0 = sc0.f[r] * scale; if (s0 + lm      >= 1000) x0 = -3.0e38f;
            float x1 = sc1.f[r] * scale; if (s0 + 16 + lm >= 1000) x1 = -3.0e38f;
            float t = fmaxf(x0, x1);
            t = fmaxf(t, __shfl_xor(t, 1));
            t = fmaxf(t, __shfl_xor(t, 2));
            t = fmaxf(t, __shfl_xor(t, 4));
            t = fmaxf(t, __shfl_xor(t, 8));
            float mn = fmaxf(m[r], t);
            float al = __expf(m[r] - mn);
            float e0 = __expf(x0 - mn);
            float e1 = __expf(x1 - mn);
            float ps = e0 + e1;
            ps += __shfl_xor(ps, 1);
            ps += __shfl_xor(ps, 2);
            ps += __shfl_xor(ps, 4);
            ps += __shfl_xor(ps, 8);
            lsum[r] = lsum[r] * al + ps;
            m[r] = mn;
            o.f[r] *= al;
            p0[r] = e0; p1[r] = e1;
        }

        // D-layout -> A-layout exchange for P through per-wave LDS (1 KB).
        asm volatile("s_wait_dscnt 0" ::: "memory");  // prior reads done before overwrite
        BF16x16 cv;
        #pragma unroll
        for (int r = 0; r < 8; ++r) {
            int row = r + half * 8;
            cv.h[0] = f2bf(p0[r]); Pex[wave][row][lm]      = cv.u[0];
            cv.h[1] = f2bf(p1[r]); Pex[wave][row][16 + lm] = cv.u[1];
        }
        asm volatile("s_wait_dscnt 0" ::: "memory");  // writes visible before reads
        BF16x16 ap;
        {
            const unsigned int* Lr = (const unsigned int*)&Pex[wave][lm][0];
            #pragma unroll
            for (int j = 0; j < 4; ++j) ap.d[j]     = Lr[half * 4 + j];       // K=half*8+2j
            #pragma unroll
            for (int j = 0; j < 4; ++j) ap.d[4 + j] = Lr[8 + half * 4 + j];   // K=16+half*8+2j
        }

        // V as B-matrix: two 16B loads per lane (K=half*8+j, 16+half*8+j) from transposed Vt[h][e][s].
        BF16x16 bv;
        const unsigned short* vp = Vt + ((size_t)h * 16 + lm) * 1024 + s0 + half * 8;
        *(uint4*)&bv.d[0] = ((const uint4*)vp)[0];
        *(uint4*)&bv.d[4] = *(const uint4*)(vp + 16);

        o.v = wmma_bf16(ap, bv, o.v);
    }

    BF16x16 cv;
    #pragma unroll
    for (int r = 0; r < 8; ++r) {
        cv.h[0] = f2bf(o.f[r] / lsum[r]);
        int row = l0 + r + half * 8;
        Ob[((size_t)(b * 1024 + row)) * 128 + h * 16 + lm] = cv.u[0];
    }
}

// ----------------------------- output projection ---------------------------
__global__ __launch_bounds__(128) void proj_o_k(
    const unsigned short* __restrict__ Ob, const unsigned short* __restrict__ WoT,
    const float* __restrict__ bo, float* __restrict__ out) {
    int lane = threadIdx.x & 31, wave = threadIdx.x >> 5;
    int half = lane >> 4, lm = lane & 15;
    int tile = blockIdx.x * 4 + wave;       // 0..65535
    int mt   = tile >> 5, nt = tile & 31;
    int row  = mt * 16 + lm;

    v8f acc = {};
    const unsigned short* bp = WoT + (size_t)(nt * 16 + lm) * 128 + half * 16;
    for (int k0 = 0; k0 < 128; k0 += 32) {
        BF16x16 a, b;
        const unsigned short* ar = Ob + (size_t)row * 128 + k0 + half * 16;
        *(uint4*)&a.d[0] = *(const uint4*)ar;
        *(uint4*)&a.d[4] = *(const uint4*)(ar + 8);
        *(uint4*)&b.d[0] = ((const uint4*)(bp + k0))[0];
        *(uint4*)&b.d[4] = ((const uint4*)(bp + k0))[1];
        acc = wmma_bf16(a, b, acc);
    }
    F32x8 d; d.v = acc;
    float bb = bo[nt * 16 + lm];
    #pragma unroll
    for (int r = 0; r < 8; ++r)
        out[((size_t)mt * 16 + r + half * 8) * 512 + nt * 16 + lm] = d.f[r] + bb;
}

// ------------------------------- launcher ----------------------------------
extern "C" void kernel_launch(void* const* d_in, const int* in_sizes, int n_in,
                              void* d_out, int out_size, void* d_ws, size_t ws_size,
                              hipStream_t stream) {
    const float* tgt = (const float*)d_in[0];
    const float* src = (const float*)d_in[1];
    const float* val = (const float*)d_in[2];
    const float* Wq  = (const float*)d_in[3];
    const float* bq  = (const float*)d_in[4];
    const float* Wk  = (const float*)d_in[5];
    const float* bk  = (const float*)d_in[6];
    const float* Wv  = (const float*)d_in[7];
    const float* bv  = (const float*)d_in[8];
    const float* Wo  = (const float*)d_in[9];
    const float* bo  = (const float*)d_in[10];
    float* out = (float*)d_out;

    // Workspace (bf16 elements): Qb 8MB | Ob 8MB | Kb 256KB | Vt 256KB |
    // WqT 128KB | WkT 1MB | WvT 1MB | WoT 128KB   (~18.7MB total)
    unsigned short* Qb  = (unsigned short*)d_ws;
    unsigned short* Ob  = Qb  + (size_t)32768 * 128;
    unsigned short* Kb  = Ob  + (size_t)32768 * 128;
    unsigned short* Vt  = Kb  + (size_t)8 * 1024 * 16;
    unsigned short* WqT = Vt  + (size_t)8 * 1024 * 16;
    unsigned short* WkT = WqT + (size_t)128 * 512;
    unsigned short* WvT = WkT + (size_t)128 * 4096;
    unsigned short* WoT = WvT + (size_t)128 * 4096;

    prepack_w_k<<<256,  256, 0, stream>>>(Wq, WqT, 512, 128);
    prepack_w_k<<<2048, 256, 0, stream>>>(Wk, WkT, 4096, 128);
    prepack_w_k<<<2048, 256, 0, stream>>>(Wv, WvT, 4096, 128);
    prepack_w_k<<<256,  256, 0, stream>>>(Wo, WoT, 128, 512);

    proj_kv_k<<<256,   128, 0, stream>>>(src, val, WkT, bk, WvT, bv, Kb, Vt);
    proj_q_k <<<4096,  128, 0, stream>>>(tgt, WqT, bq, Qb);
    attn_k   <<<4096,  128, 0, stream>>>(Qb, Kb, Vt, Ob);
    proj_o_k <<<16384, 128, 0, stream>>>(Ob, WoT, bo, out);
}
